// MutualInformationLoss_2997887173181
// MI455X (gfx1250) — matrix operand order repaired and checked by reading, not compiled, not run on hardware
//
#include <hip/hip_runtime.h>
#include <hip/hip_bf16.h>

// ---------------- problem constants (from reference) ----------------
#define NBINS    64
#define NBATCH   4
#define NPIX     262144          // 512*512 pixels per batch image
#define EPSV     1e-10f

// ---------------- kernel geometry ----------------
#define BPB      128             // blocks per batch
#define PIXPB    (NPIX / BPB)    // 2048 pixels per block
#define CHUNK    512             // pixels staged in LDS per iteration
#define ITERS    (PIXPB / CHUNK) // 4
#define PITCH    (CHUNK + 8)     // 520 halves: pad => conflict-free frag loads
#define NTHREADS 256             // 8 wave32
#define NWAVES   8

typedef _Float16 h2   __attribute__((ext_vector_type(2)));
typedef _Float16 h8   __attribute__((ext_vector_type(8)));
typedef _Float16 v16h __attribute__((ext_vector_type(16)));
typedef float    v8f  __attribute__((ext_vector_type(8)));

// Soft-bin weights for two pixels, stored packed-f16 into bin-major LDS:
//   w_i(x) = exp(-2(x-c_i)^2) / sum_i,  c_i = i/63
// Table-free double recurrence (no LDS reads):
//   e_i = A * r^i * exp(-2 c_i^2);  e_{i+1} = e_i * u_i;  u_{i+1} = u_i * Q
//   A = exp(-2x^2), r = exp(4hx), C = exp(-2h^2), Q = exp(-4h^2), u_0 = r*C
__device__ __forceinline__ void stage_pair(_Float16* wT, int col,
                                           float x0, float x1,
                                           float C, float Q) {
  const float h = 1.0f / 63.0f;
  x0 = fminf(fmaxf(x0, 0.0f), 1.0f);
  x1 = fminf(fmaxf(x1, 0.0f), 1.0f);
  const float r0 = __expf(4.0f * h * x0), A0 = __expf(-2.0f * x0 * x0);
  const float r1 = __expf(4.0f * h * x1), A1 = __expf(-2.0f * x1 * x1);

  // pass 1: sums
  float e0 = A0, u0 = r0 * C;
  float e1 = A1, u1 = r1 * C;
  float s0 = 0.0f, s1 = 0.0f;
  #pragma unroll
  for (int i = 0; i < NBINS; ++i) {
    s0 += e0; s1 += e1;
    e0 *= u0; e1 *= u1;
    u0 *= Q;  u1 *= Q;
  }
  const float i0 = 1.0f / (s0 + EPSV);
  const float i1 = 1.0f / (s1 + EPSV);

  // pass 2: normalized f16 stores (normalization folded into e)
  e0 = A0 * i0; u0 = r0 * C;
  e1 = A1 * i1; u1 = r1 * C;
  #pragma unroll
  for (int i = 0; i < NBINS; ++i) {
    h2 hv;
    hv.x = (_Float16)e0;
    hv.y = (_Float16)e1;
    *(h2*)(wT + i * PITCH + col) = hv;  // ds_store_b32, packed pixel pair
    e0 *= u0; e1 *= u1;
    u0 *= Q;  u1 *= Q;
  }
}

// Kernel 1: per-block partial joint histogram via WMMA.
__global__ __launch_bounds__(NTHREADS, 1)
void mi_joint_kernel(const float* __restrict__ fixedI,
                     const float* __restrict__ movingI,
                     float* __restrict__ blockPartial) {
  __shared__ __align__(16) _Float16 stage[2 * NBINS * PITCH]; // 133,120 B

  _Float16* wxT = stage;                    // [64][PITCH] bins of fixed
  _Float16* wyT = stage + NBINS * PITCH;    // [64][PITCH] bins of moving
  float* redBuf = (float*)stage;            // reused after final barrier

  const int tid  = threadIdx.x;
  const int lane = tid & 31;
  const int wv   = tid >> 5;
  const int b    = blockIdx.x / BPB;
  const int blk  = blockIdx.x % BPB;
  const int m16  = lane & 15;
  const int grp  = lane >> 4;

  const float h = 1.0f / 63.0f;
  const float C = __expf(-2.0f * h * h);
  const float Q = __expf(-4.0f * h * h);

  v8f acc[4][4];
  #pragma unroll
  for (int t = 0; t < 4; ++t)
    #pragma unroll
    for (int u = 0; u < 4; ++u)
      #pragma unroll
      for (int r = 0; r < 8; ++r) acc[t][u][r] = 0.0f;

  const int pixBase = b * NPIX + blk * PIXPB;

  for (int it = 0; it < ITERS; ++it) {
    __syncthreads();  // staging safe to overwrite

    // ---- stage 512 pixels of soft weights (2 pixels / thread) ----
    const int col = 2 * tid;
    const int gp  = pixBase + it * CHUNK + col;
    const float2 fx2 = *(const float2*)(fixedI  + gp);
    const float2 mv2 = *(const float2*)(movingI + gp);
    stage_pair(wxT, col, fx2.x, fx2.y, C, Q);
    stage_pair(wyT, col, mv2.x, mv2.y, C, Q);
    __syncthreads();

    // ---- WMMA: this wave accumulates K range [64*wv, 64*wv+64) ----
    #pragma unroll
    for (int s = 0; s < 2; ++s) {
      const int kb = wv * 64 + 32 * s;

      v16h bf[4];
      #pragma unroll
      for (int u = 0; u < 4; ++u) {
        // 16-bit B 32x16 layout: lane n=l&15; halves 0-7: K=8g..8g+7,
        // halves 8-15: K=16+8g..16+8g+7
        const _Float16* bp = &wyT[(16 * u + m16) * PITCH + kb + 8 * grp];
        const h8 lo = *(const h8*)bp;
        const h8 hi = *(const h8*)(bp + 16);
        #pragma unroll
        for (int i = 0; i < 8; ++i) { bf[u][i] = lo[i]; bf[u][8 + i] = hi[i]; }
      }

      #pragma unroll
      for (int t = 0; t < 4; ++t) {
        // 16-bit A 16x32 layout: lane m=l&15; halves 0-7: K=8g..8g+7,
        // halves 8-15: K=16+8g..16+8g+7
        const _Float16* ap = &wxT[(16 * t + m16) * PITCH + kb + 8 * grp];
        const h8 lo = *(const h8*)ap;
        const h8 hi = *(const h8*)(ap + 16);
        v16h af;
        #pragma unroll
        for (int i = 0; i < 8; ++i) { af[i] = lo[i]; af[8 + i] = hi[i]; }

        #pragma unroll
        for (int u = 0; u < 4; ++u) {
          acc[t][u] = __builtin_amdgcn_wmma_f32_16x16x32_f16(
              false, af, false, bf[u], (short)0, acc[t][u], false, false);
        }
      }
    }
  }

  // ---- deterministic in-block reduction of 8 per-wave 64x64 partials ----
  __syncthreads();
  {
    float* myR = redBuf + wv * 4096;
    #pragma unroll
    for (int t = 0; t < 4; ++t)
      #pragma unroll
      for (int u = 0; u < 4; ++u)
        #pragma unroll
        for (int r = 0; r < 8; ++r) {
          const int row = 16 * t + r + 8 * grp;   // C/D layout: VGPR r -> M
          const int cc  = 16 * u + m16;           // lane -> N
          myR[row * 64 + cc] = acc[t][u][r];
        }
  }
  __syncthreads();

  float* outP = blockPartial + (size_t)blockIdx.x * 4096;
  for (int e = tid; e < 4096; e += NTHREADS) {
    float s = 0.0f;
    #pragma unroll
    for (int w8 = 0; w8 < NWAVES; ++w8) s += redBuf[w8 * 4096 + e];
    outP[e] = s;
  }
}

// Kernel 2: deterministic fold of per-block partials -> per-batch joint.
__global__ void mi_reduce_kernel(const float* __restrict__ partial,
                                 float* __restrict__ jointG) {
  const int idx = blockIdx.x * blockDim.x + threadIdx.x;  // 0..16383
  const int b = idx >> 12;
  const int e = idx & 4095;
  const float* p = partial + (size_t)b * BPB * 4096 + e;
  float s = 0.0f;
  for (int k = 0; k < BPB; ++k) s += p[(size_t)k * 4096];
  jointG[idx] = s;
}

// Kernel 3: entropies + MI from the 4 joint histograms (single block).
__global__ __launch_bounds__(256, 1)
void mi_loss_kernel(const float* __restrict__ jointG, float* __restrict__ out) {
  __shared__ float J[4096];
  __shared__ float rs[64], cs[64], red[256];
  __shared__ float Sshare, miAcc;
  const int tid = threadIdx.x;
  if (tid == 0) miAcc = 0.0f;

  for (int b = 0; b < NBATCH; ++b) {
    __syncthreads();
    for (int e = tid; e < 4096; e += 256) J[e] = jointG[b * 4096 + e];
    __syncthreads();

    if (tid < 64) {
      float r = 0.0f, c = 0.0f;
      for (int i = 0; i < 64; ++i) {
        r += J[tid * 64 + i];
        c += J[i * 64 + tid];
      }
      rs[tid] = r; cs[tid] = c;
    }
    __syncthreads();
    if (tid == 0) {
      float S = 0.0f;
      for (int i = 0; i < 64; ++i) S += rs[i];
      Sshare = S;
    }
    __syncthreads();

    const float invS = 1.0f / (Sshare + EPSV);
    float hj = 0.0f;
    for (int e = tid; e < 4096; e += 256) {
      const float pn = J[e] * invS + EPSV;
      hj += pn * __logf(pn);
    }
    red[tid] = hj;
    __syncthreads();

    if (tid == 0) {
      float Hj = 0.0f;
      for (int i = 0; i < 256; ++i) Hj += red[i];
      Hj = -Hj;
      float Hx = 0.0f, Hy = 0.0f;
      for (int i = 0; i < 64; ++i) {
        const float px = rs[i] * invS + EPSV;
        const float py = cs[i] * invS + EPSV;
        Hx -= px * __logf(px);
        Hy -= py * __logf(py);
      }
      miAcc += (Hx + Hy - Hj);
    }
  }
  __syncthreads();
  if (tid == 0) out[0] = -miAcc * (1.0f / (float)NBATCH);
}

extern "C" void kernel_launch(void* const* d_in, const int* in_sizes, int n_in,
                              void* d_out, int out_size, void* d_ws, size_t ws_size,
                              hipStream_t stream) {
  (void)in_sizes; (void)n_in; (void)out_size; (void)ws_size;
  const float* fixedI  = (const float*)d_in[0];
  const float* movingI = (const float*)d_in[1];

  float* partial = (float*)d_ws;                               // [512][4096]
  float* jointG  = partial + (size_t)NBATCH * BPB * 4096;      // [4][4096]

  mi_joint_kernel<<<NBATCH * BPB, NTHREADS, 0, stream>>>(fixedI, movingI, partial);
  mi_reduce_kernel<<<(NBATCH * 4096) / 256, 256, 0, stream>>>(partial, jointG);
  mi_loss_kernel<<<1, 256, 0, stream>>>(jointG, (float*)d_out);
}
